// Corssattention_47132971106683
// MI455X (gfx1250) — matrix operand order repaired and checked by reading, not compiled, not run on hardware
//
#include <hip/hip_runtime.h>
#include <math.h>

// ---------------------------------------------------------------------------
// Problem constants (match reference)
// ---------------------------------------------------------------------------
#define B_DIM   8
#define C_DIM   256
#define H_DIM   64
#define W_DIM   64
#define N_PIX   (H_DIM * W_DIM)      // 4096
#define HID_DIM (4 * C_DIM)          // 1024
#define M_DIM   (C_DIM / 8)          // 32
#define EPSF    1e-6f

typedef __attribute__((ext_vector_type(16))) _Float16 v16h;
typedef __attribute__((ext_vector_type(8)))  _Float16 h8;
typedef __attribute__((ext_vector_type(4)))  _Float16 h4;
typedef __attribute__((ext_vector_type(8)))  float    v8f;
typedef __attribute__((ext_vector_type(4)))  float    f4;

union H8  { h8  v; _Float16 h[8];  };
union H4  { h4  v; _Float16 h[4];  };
union V16 { v16h v; h8 p[2]; };

// ---------------------------------------------------------------------------
// WMMA GEMM:  Cout[M,N] = epilogue( A[M,K] * B[K,N] )
//   BT = 0 : B stored [K][Ncols] row-major
//   BT = 1 : B stored [Ncols][K] row-major (multiply by transpose)
// Tile: 64(M) x 128(N) per 256-thread block, K step 32.
// 8 waves; wave owns 16(M) x 64(N) -> 4 f32 accumulators, A frag reused 4x.
// LDS layouts are fragment-friendly: every LDS access is b64/b128.
//   As[row][k]  pitch 40 halves : A frag = 2 contiguous 8-half runs per lane
//   Bs[col][k]  pitch 40 halves : B frag = 2 contiguous 8-half runs per lane
// Requirements: Ncols % 128 == 0, K % 32 == 0 (true for all call sites).
// Epilogue: +bias[row], *bn_s[row]+bn_t[row], act (1=relu6, 2=softplus),
//           *(*alpha_ptr), + addsrc[row][col].
// ---------------------------------------------------------------------------
template <int BT>
__global__ __launch_bounds__(256)
void gemm_wmma(const float* __restrict__ A, const float* __restrict__ Bm,
               float* __restrict__ Cout,
               int Mrows, int Ncols, int K,
               long strideA, long strideB, long strideC,
               const float* __restrict__ bias,
               const float* __restrict__ bns, const float* __restrict__ bnt,
               int act,
               const float* __restrict__ alpha_ptr,
               const float* __restrict__ addsrc, long strideAdd)
{
    const int AP = 40;                 // A LDS pitch in halves (80B, 16B-mult)
    const int BP = 40;                 // B LDS pitch in halves
    __shared__ _Float16 As[64 * 40];
    __shared__ _Float16 Bs[128 * 40];

    const int z = blockIdx.z;
    const float* Ab = A  + (long)z * strideA;
    const float* Bb = Bm + (long)z * strideB;
    float*       Cb = Cout + (long)z * strideC;

    const int tid  = threadIdx.x;
    const int lane = tid & 31;
    const int wave = tid >> 5;

    const int Mbase = blockIdx.y * 64;
    const int Nbase = blockIdx.x * 128;

    // ---- A staging: thread -> (row, 8-wide k chunk), branch-free row guard
    const int   sa_r   = tid >> 2;            // 0..63
    const int   sa_e   = (tid & 3) * 8;       // 0,8,16,24
    const int   sa_gr  = Mbase + sa_r;
    const int   sa_grc = (sa_gr < Mrows) ? sa_gr : (Mrows - 1);
    const float sa_msk = (sa_gr < Mrows) ? 1.0f : 0.0f;

    // ---- fragment addressing (ISA 16-bit A 16x32 / B 32x16 lane layouts)
    const int frow = (wave & 3) * 16 + (lane & 15);
    const int fkh  = (lane & 16) ? 8 : 0;     // A: elems 0-7 at K=kh.., 8-15 at K=kh+16..
    const int fkb  = (lane & 16) ? 8 : 0;     // B: elems 0-7 at K=kb.., 8-15 at K=kb+16..
    const int fcb  = (wave >> 2) * 64 + (lane & 15);

    v8f acc[4] = { {}, {}, {}, {} };

    for (int k0 = 0; k0 < K; k0 += 32) {
        // -------- stage A tile 64x32 : 2x b128 global, 1x b128 LDS store ----
        {
            const float* ap = Ab + (long)sa_grc * K + k0 + sa_e;
            f4 a0 = *(const f4*)ap;
            f4 a1 = *(const f4*)(ap + 4);
            H8 hv;
            #pragma unroll
            for (int i = 0; i < 4; ++i) {
                hv.h[i]     = (_Float16)(a0[i] * sa_msk);
                hv.h[4 + i] = (_Float16)(a1[i] * sa_msk);
            }
            *(h8*)&As[sa_r * AP + sa_e] = hv.v;
        }
        // -------- stage B tile 32x128 (transposed into LDS) -----------------
        if (BT) {
            // global rows are K-contiguous: thread = (col, 16-wide k chunk)
            int c  = tid & 127;
            int k8 = (tid >> 7) * 16;
            const float* bp = Bb + (long)(Nbase + c) * K + k0 + k8;
            f4 b0 = *(const f4*)bp;
            f4 b1 = *(const f4*)(bp + 4);
            f4 b2 = *(const f4*)(bp + 8);
            f4 b3 = *(const f4*)(bp + 12);
            H8 lo, hi;
            #pragma unroll
            for (int i = 0; i < 4; ++i) {
                lo.h[i]     = (_Float16)b0[i];
                lo.h[4 + i] = (_Float16)b1[i];
                hi.h[i]     = (_Float16)b2[i];
                hi.h[4 + i] = (_Float16)b3[i];
            }
            *(h8*)&Bs[c * BP + k8]     = lo.v;
            *(h8*)&Bs[c * BP + k8 + 8] = hi.v;
        } else {
            // global rows are N-contiguous: 4(col) x 4(k) micro-tile / thread
            int c0 = (tid & 31) * 4;
            int k4 = (tid >> 5) * 4;
            H4 col[4];
            #pragma unroll
            for (int i = 0; i < 4; ++i) {
                f4 v = *(const f4*)(Bb + (long)(k0 + k4 + i) * Ncols + Nbase + c0);
                #pragma unroll
                for (int j = 0; j < 4; ++j) col[j].h[i] = (_Float16)v[j];
            }
            #pragma unroll
            for (int j = 0; j < 4; ++j)
                *(h4*)&Bs[(c0 + j) * BP + k4] = col[j].v;
        }
        __syncthreads();

        // speculative prefetch of next A chunk (gfx1250 global_prefetch_b8)
        if (k0 + 32 < K && sa_gr < Mrows)
            __builtin_prefetch(Ab + (long)sa_gr * K + k0 + 32 + sa_e, 0, 1);

        // -------- fragments: all b128 LDS loads -----------------------------
        V16 a;
        a.p[0] = *(const h8*)&As[frow * AP + fkh];
        a.p[1] = *(const h8*)&As[frow * AP + fkh + 16];
        #pragma unroll
        for (int j = 0; j < 4; ++j) {
            V16 bf;
            const int co = (fcb + j * 16) * BP + fkb;
            bf.p[0] = *(const h8*)&Bs[co];
            bf.p[1] = *(const h8*)&Bs[co + 16];
            acc[j] = __builtin_amdgcn_wmma_f32_16x16x32_f16(
                false, a.v, false, bf.v, (short)0, acc[j], false, false);
        }
        __syncthreads();
    }

    // -------- epilogue (branch-free per element) ----------------------------
    const float  alpha = alpha_ptr ? alpha_ptr[0] : 1.0f;
    const float* addb  = addsrc ? (addsrc + (long)z * strideAdd) : (const float*)0;

    const int rbase = Mbase + (wave & 3) * 16 + ((lane & 16) ? 8 : 0);
    const int cbase = Nbase + (wave >> 2) * 64 + (lane & 15);

    #pragma unroll
    for (int r = 0; r < 8; ++r) {
        int row = rbase + r;
        if (row >= Mrows) continue;
        const float bi = bias ? bias[row] : 0.0f;
        const float ss = bns ? bns[row] : 1.0f;
        const float tt = bns ? bnt[row] : 0.0f;
        #pragma unroll
        for (int j = 0; j < 4; ++j) {
            float v = acc[j][r] + bi;
            v = v * ss + tt;
            if (act == 1)      v = fminf(fmaxf(v, 0.0f), 6.0f);
            else if (act == 2) v = (v > 20.0f) ? v : log1pf(__expf(v));
            v *= alpha;
            const int cc = cbase + j * 16;
            if (addb) v += addb[(long)row * Ncols + cc];
            Cb[(long)row * Ncols + cc] = v;
        }
    }
}

// ---------------------------------------------------------------------------
// depthwise 3x3 (pad=1) + folded conv-BN + folded norm-BN
// ---------------------------------------------------------------------------
__global__ void dw3_bn2(const float* __restrict__ in, float* __restrict__ out,
                        const float* __restrict__ dw,
                        const float* __restrict__ cbs, const float* __restrict__ cbt,
                        const float* __restrict__ ns,  const float* __restrict__ nt)
{
    long idx = (long)blockIdx.x * blockDim.x + threadIdx.x;
    if (idx >= (long)B_DIM * C_DIM * N_PIX) return;
    int w = idx & (W_DIM - 1);
    long t = idx >> 6;
    int h = t & (H_DIM - 1); t >>= 6;
    int c = t & (C_DIM - 1);
    int b = (int)(t >> 8);
    const float* base = in + (((long)b * C_DIM + c) * N_PIX);
    const float* k    = dw + c * 9;
    float s = 0.0f;
    #pragma unroll
    for (int dy = -1; dy <= 1; ++dy)
        #pragma unroll
        for (int dx = -1; dx <= 1; ++dx) {
            int hh = h + dy, ww = w + dx;
            if (hh >= 0 && hh < H_DIM && ww >= 0 && ww < W_DIM)
                s += k[(dy + 1) * 3 + (dx + 1)] * base[hh * W_DIM + ww];
        }
    s = s * cbs[c] + cbt[c];
    s = s * ns[c] + nt[c];
    out[idx] = s;
}

// KV[b][m][c] = sum_n K[b][m][n] * V[b][c][n]        grid (C, M, B), 256 thr
__global__ void kv_reduce(const float* __restrict__ Kmat,
                          const float* __restrict__ Vmat,
                          float* __restrict__ KV)
{
    int c = blockIdx.x, m = blockIdx.y, b = blockIdx.z;
    const float* kr = Kmat + ((long)b * M_DIM + m) * N_PIX;
    const float* vr = Vmat + ((long)b * C_DIM + c) * N_PIX;
    float s = 0.0f;
    for (int n = threadIdx.x; n < N_PIX; n += 256) s += kr[n] * vr[n];
    __shared__ float red[256];
    red[threadIdx.x] = s;
    __syncthreads();
    for (int off = 128; off > 0; off >>= 1) {
        if (threadIdx.x < off) red[threadIdx.x] += red[threadIdx.x + off];
        __syncthreads();
    }
    if (threadIdx.x == 0) KV[((long)b * M_DIM + m) * C_DIM + c] = red[0];
}

// ksum[b][m] = sum_n K[b][m][n]                        grid (M, B), 256 thr
__global__ void ksum_reduce(const float* __restrict__ Kmat, float* __restrict__ ksum)
{
    int m = blockIdx.x, b = blockIdx.y;
    const float* kr = Kmat + ((long)b * M_DIM + m) * N_PIX;
    float s = 0.0f;
    for (int n = threadIdx.x; n < N_PIX; n += 256) s += kr[n];
    __shared__ float red[256];
    red[threadIdx.x] = s;
    __syncthreads();
    for (int off = 128; off > 0; off >>= 1) {
        if (threadIdx.x < off) red[threadIdx.x] += red[threadIdx.x + off];
        __syncthreads();
    }
    if (threadIdx.x == 0) ksum[(long)b * M_DIM + m] = red[0];
}

// inv[b][n] = 1 / sum_m Q[b][m][n] * (ksum[b][m] + eps)   grid (N/256, B)
__global__ void inv_kernel(const float* __restrict__ Q,
                           const float* __restrict__ ksum,
                           float* __restrict__ inv)
{
    int b = blockIdx.y;
    int n = blockIdx.x * 256 + threadIdx.x;
    float s = 0.0f;
    #pragma unroll
    for (int m = 0; m < M_DIM; ++m)
        s += Q[((long)b * M_DIM + m) * N_PIX + n] * (ksum[(long)b * M_DIM + m] + EPSF);
    inv[(long)b * N_PIX + n] = 1.0f / s;
}

// out[b][c][n] = 2*z + gamma_k * inv[n] * sum_m Q[m][n]*KV[m][c]
// grid (N/256, C, B)
__global__ void wv_kernel(const float* __restrict__ Q,
                          const float* __restrict__ KV,
                          const float* __restrict__ inv,
                          const float* __restrict__ z,
                          const float* __restrict__ gamma,
                          float* __restrict__ out)
{
    int b = blockIdx.z, c = blockIdx.y;
    int n = blockIdx.x * 256 + threadIdx.x;
    float s = 0.0f;
    #pragma unroll
    for (int m = 0; m < M_DIM; ++m)
        s += Q[((long)b * M_DIM + m) * N_PIX + n] * KV[((long)b * M_DIM + m) * C_DIM + c];
    long idx = ((long)b * C_DIM + c) * N_PIX + n;
    out[idx] = 2.0f * z[idx] + gamma[0] * inv[(long)b * N_PIX + n] * s;
}

// att = softmax(rowmax(E) - E) along last dim, in place.  grid (C, B), 256 thr
__global__ void catt_softmax(float* __restrict__ energy)
{
    int c = blockIdx.x, b = blockIdx.y;
    float* row = energy + ((long)b * C_DIM + c) * C_DIM;
    int d = threadIdx.x;
    float e = row[d];
    __shared__ float red[256];

    red[d] = e; __syncthreads();
    for (int off = 128; off > 0; off >>= 1) {
        if (d < off) red[d] = fmaxf(red[d], red[d + off]);
        __syncthreads();
    }
    float rowmax = red[0];
    __syncthreads();

    float arg = rowmax - e;
    red[d] = arg; __syncthreads();
    for (int off = 128; off > 0; off >>= 1) {
        if (d < off) red[d] = fmaxf(red[d], red[d + off]);
        __syncthreads();
    }
    float amax = red[0];
    __syncthreads();

    float ex = __expf(arg - amax);
    red[d] = ex; __syncthreads();
    for (int off = 128; off > 0; off >>= 1) {
        if (d < off) red[d] += red[d + off];
        __syncthreads();
    }
    row[d] = ex / red[0];
}

// out = (a * b) * ns[c] + nt[c]
__global__ void mul_norm(const float* __restrict__ a, const float* __restrict__ bm,
                         const float* __restrict__ ns, const float* __restrict__ nt,
                         float* __restrict__ out)
{
    long idx = (long)blockIdx.x * blockDim.x + threadIdx.x;
    if (idx >= (long)B_DIM * C_DIM * N_PIX) return;
    int c = (int)((idx >> 12) & (C_DIM - 1));
    out[idx] = (a[idx] * bm[idx]) * ns[c] + nt[c];
}

// y3 = y + y2 ; x3 = x + x2 + y3 ; out = [x3 | y3]
__global__ void final_combine(const float* __restrict__ x, const float* __restrict__ y,
                              const float* __restrict__ x2, const float* __restrict__ y2,
                              float* __restrict__ out)
{
    long idx = (long)blockIdx.x * blockDim.x + threadIdx.x;
    const long F = (long)B_DIM * C_DIM * N_PIX;
    if (idx >= F) return;
    float y3 = y[idx] + y2[idx];
    out[F + idx] = y3;
    out[idx]     = x[idx] + x2[idx] + y3;
}

// ---------------------------------------------------------------------------
// Host orchestration
// ---------------------------------------------------------------------------
static void launch_gemm(bool bt,
                        const float* A, const float* Bm, float* Cm,
                        int Mr, int Nc, int K,
                        long sA, long sB, long sC, int nb,
                        const float* bias, const float* bns, const float* bnt,
                        int act, const float* alpha, const float* add, long sAdd,
                        hipStream_t stream)
{
    dim3 g(Nc / 128, (Mr + 63) / 64, nb);
    if (bt)
        gemm_wmma<1><<<g, 256, 0, stream>>>(A, Bm, Cm, Mr, Nc, K, sA, sB, sC,
                                            bias, bns, bnt, act, alpha, add, sAdd);
    else
        gemm_wmma<0><<<g, 256, 0, stream>>>(A, Bm, Cm, Mr, Nc, K, sA, sB, sC,
                                            bias, bns, bnt, act, alpha, add, sAdd);
}

extern "C" void kernel_launch(void* const* d_in, const int* in_sizes, int n_in,
                              void* d_out, int out_size, void* d_ws, size_t ws_size,
                              hipStream_t stream)
{
    const float* x       = (const float*)d_in[0];
    const float* y       = (const float*)d_in[1];
    const float* pw_w    = (const float*)d_in[2];
    const float* dw_w    = (const float*)d_in[3];
    const float* cbn_s   = (const float*)d_in[4];
    const float* cbn_t   = (const float*)d_in[5];
    const float* norm_s  = (const float*)d_in[6];
    const float* norm_t  = (const float*)d_in[7];
    const float* q_w     = (const float*)d_in[8];
    const float* k_w     = (const float*)d_in[9];
    const float* v_w     = (const float*)d_in[10];
    const float* gamma_k = (const float*)d_in[11];
    const float* gamma_c = (const float*)d_in[12];
    const float* fc1_w   = (const float*)d_in[13];
    const float* fc1_b   = (const float*)d_in[14];
    const float* bn1_s   = (const float*)d_in[15];
    const float* bn1_t   = (const float*)d_in[16];
    const float* fc2_w   = (const float*)d_in[17];
    const float* fc2_b   = (const float*)d_in[18];
    const float* bn2_s   = (const float*)d_in[19];
    const float* bn2_t   = (const float*)d_in[20];

    const long F  = (long)B_DIM * C_DIM * N_PIX;      // 8.39M floats per tensor
    const long CN = (long)C_DIM * N_PIX;

    float* ws   = (float*)d_ws;
    float* buf0 = ws;                                  // pw scratch / V / mlp input
    float* x1   = buf0 + F;
    float* y1   = x1 + F;
    float* lx   = y1 + F;                              // light(x1)
    float* ly   = lx + F;                              // light(y1)
    float* hid  = ly + F;                              // HID*N (one image)
    float* qb   = hid + (long)HID_DIM * N_PIX;         // B*M*N
    float* kb   = qb + (long)B_DIM * M_DIM * N_PIX;
    float* kvb  = kb + (long)B_DIM * M_DIM * N_PIX;    // B*M*C
    float* ksb  = kvb + (long)B_DIM * M_DIM * C_DIM;   // B*M
    float* invb = ksb + (long)B_DIM * M_DIM;           // B*N
    float* ebuf = invb + (long)B_DIM * N_PIX;          // B*C*C

    const int EW_BLOCKS = (int)((F + 255) / 256);

    // ---- stem: x1 = norm(sepconvbn(x)), y1 = norm(sepconvbn(y)) ----
    launch_gemm(false, pw_w, x, buf0, C_DIM, N_PIX, C_DIM, 0, CN, CN, B_DIM,
                0, 0, 0, 0, 0, 0, 0, stream);
    dw3_bn2<<<EW_BLOCKS, 256, 0, stream>>>(buf0, x1, dw_w, cbn_s, cbn_t, norm_s, norm_t);

    launch_gemm(false, pw_w, y, buf0, C_DIM, N_PIX, C_DIM, 0, CN, CN, B_DIM,
                0, 0, 0, 0, 0, 0, 0, stream);
    dw3_bn2<<<EW_BLOCKS, 256, 0, stream>>>(buf0, y1, dw_w, cbn_s, cbn_t, norm_s, norm_t);

    // ---- light(z) = 2z + gamma_k*wv + gamma_c*att@z  for z in {x1, y1} ----
    const float* zs[2]   = { x1, y1 };
    float*       outs[2] = { lx, ly };
    for (int p = 0; p < 2; ++p) {
        const float* z = zs[p];
        float* lz = outs[p];
        const long MN = (long)M_DIM * N_PIX;

        // Q = softplus(q_w z), K = softplus(k_w z), V = v_w z
        launch_gemm(false, q_w, z, qb, M_DIM, N_PIX, C_DIM, 0, CN, MN, B_DIM,
                    0, 0, 0, 2, 0, 0, 0, stream);
        launch_gemm(false, k_w, z, kb, M_DIM, N_PIX, C_DIM, 0, CN, MN, B_DIM,
                    0, 0, 0, 2, 0, 0, 0, stream);
        launch_gemm(false, v_w, z, buf0, C_DIM, N_PIX, C_DIM, 0, CN, CN, B_DIM,
                    0, 0, 0, 0, 0, 0, 0, stream);

        kv_reduce<<<dim3(C_DIM, M_DIM, B_DIM), 256, 0, stream>>>(kb, buf0, kvb);
        ksum_reduce<<<dim3(M_DIM, B_DIM), 256, 0, stream>>>(kb, ksb);
        inv_kernel<<<dim3(N_PIX / 256, B_DIM), 256, 0, stream>>>(qb, ksb, invb);
        // lz = 2z + gamma_k * inv * (Q^T KV)
        wv_kernel<<<dim3(N_PIX / 256, C_DIM, B_DIM), 256, 0, stream>>>(
            qb, kvb, invb, z, gamma_k, lz);

        // catt: energy = z z^T  (B-transposed GEMM, K = N_PIX)
        launch_gemm(true, z, z, ebuf, C_DIM, C_DIM, N_PIX, CN, CN,
                    (long)C_DIM * C_DIM, B_DIM, 0, 0, 0, 0, 0, 0, 0, stream);
        catt_softmax<<<dim3(C_DIM, B_DIM), 256, 0, stream>>>(ebuf);
        // lz = gamma_c * (att @ z) + lz
        launch_gemm(false, ebuf, z, lz, C_DIM, N_PIX, C_DIM,
                    (long)C_DIM * C_DIM, CN, CN, B_DIM,
                    0, 0, 0, 0, gamma_c, lz, CN, stream);
    }

    // ---- x2 = mlp(norm(x1 * ly)), overwrite x1 ----
    mul_norm<<<EW_BLOCKS, 256, 0, stream>>>(x1, ly, norm_s, norm_t, buf0);
    for (int b = 0; b < B_DIM; ++b) {
        launch_gemm(false, fc1_w, buf0 + (long)b * CN, hid,
                    HID_DIM, N_PIX, C_DIM, 0, 0, 0, 1,
                    fc1_b, bn1_s, bn1_t, 1, 0, 0, 0, stream);
        launch_gemm(false, fc2_w, hid, x1 + (long)b * CN,
                    C_DIM, N_PIX, HID_DIM, 0, 0, 0, 1,
                    fc2_b, bn2_s, bn2_t, 0, 0, 0, 0, stream);
    }

    // ---- y2 = mlp(norm(y1 * lx)), overwrite y1 ----
    mul_norm<<<EW_BLOCKS, 256, 0, stream>>>(y1, lx, norm_s, norm_t, buf0);
    for (int b = 0; b < B_DIM; ++b) {
        launch_gemm(false, fc1_w, buf0 + (long)b * CN, hid,
                    HID_DIM, N_PIX, C_DIM, 0, 0, 0, 1,
                    fc1_b, bn1_s, bn1_t, 1, 0, 0, 0, stream);
        launch_gemm(false, fc2_w, hid, y1 + (long)b * CN,
                    C_DIM, N_PIX, HID_DIM, 0, 0, 0, 1,
                    fc2_b, bn2_s, bn2_t, 0, 0, 0, 0, stream);
    }

    // ---- y3 = y + y2 ; x3 = x + x2 + y3 ----
    final_combine<<<EW_BLOCKS, 256, 0, stream>>>(x, y, x1, y1, (float*)d_out);
}
